// LSTM_20383914787030
// MI455X (gfx1250) — matrix operand order repaired and checked by reading, not compiled, not run on hardware
//
#include <hip/hip_runtime.h>
#include <hip/hip_bf16.h>
#include <stdint.h>

typedef __bf16 bf16_t;
typedef __attribute__((ext_vector_type(16))) __bf16 v16bf;
typedef __attribute__((ext_vector_type(8)))  __bf16 v8bf;
typedef __attribute__((ext_vector_type(8)))  float  v8f;

#define B_   256
#define T_   200
#define IN_  128
#define H_   512

// LDS tile row stride (elements), padded from 32 to 40 to spread banks.
#define APAD 40

__device__ __forceinline__ float fast_sigmoid(float x) {
    return 1.0f / (1.0f + __expf(-x));
}
__device__ __forceinline__ float fast_tanh(float x) {
    x = fminf(fmaxf(x, -15.0f), 15.0f);
    float e = __expf(2.0f * x);
    return (e - 1.0f) / (e + 1.0f);
}

// CDNA5 async global->LDS DMA (no VGPR round-trip, tracked by ASYNCcnt).
// Copies 32 bytes per lane: two b128 async loads. INST_OFFSET is applied to
// both the global and the LDS address (ISA 10., async pseudocode).
__device__ __forceinline__ void async_copy32(uint32_t lds_addr, const void* gptr) {
    asm volatile(
        "global_load_async_to_lds_b128 %0, %1, off\n\t"
        "global_load_async_to_lds_b128 %0, %1, off offset:16"
        :: "v"(lds_addr), "v"(gptr) : "memory");
}
__device__ __forceinline__ void wait_async0() {
    asm volatile("s_wait_asynccnt 0x0" ::: "memory");
}
__device__ __forceinline__ uint32_t lds_addr_of(const void* p) {
    // Generic LDS pointers carry the LDS byte offset in the low 32 bits
    // (ISA 10.2 aperture mapping).
    return (uint32_t)(uintptr_t)p;
}

__global__ void cvt_f32_to_bf16(const float* __restrict__ src,
                                bf16_t* __restrict__ dst, int n) {
    int i = blockIdx.x * blockDim.x + threadIdx.x;
    int stride = gridDim.x * blockDim.x;
    for (; i < n; i += stride) dst[i] = (bf16_t)src[i];
}

__global__ void zero_u32(uint32_t* __restrict__ p, int n) {
    int i = blockIdx.x * blockDim.x + threadIdx.x;
    int stride = gridDim.x * blockDim.x;
    for (; i < n; i += stride) p[i] = 0u;
}

// One fused LSTM timestep:
//   gates[m, g*512+j] = [x_t | h_prev](m, :) . [Wih | Whh](g*512+j, :)
//   + biases, activations, cell update, h write (bf16), optional h_all write.
// Grid: (B/64, H/64), block 128 threads (4 waves, each wave a 32x32 D tile
// replicated over the 4 gate blocks). Double-buffered LDS tiles filled via
// async global->LDS DMA; one barrier per K-step.
__global__ __launch_bounds__(128) void lstm_step_kernel(
    const bf16_t* __restrict__ xp, long xstride, int KA,  // A cols [0, KA)
    const bf16_t* __restrict__ hprev,                     // [B][H] bf16
    const bf16_t* __restrict__ Wih,                       // [4H][KA] bf16
    const bf16_t* __restrict__ Whh,                       // [4H][H] bf16
    const float*  __restrict__ bih,
    const float*  __restrict__ bhh,
    float*        __restrict__ cst,                       // [B][H] f32 state
    bf16_t*       __restrict__ hout,                      // [B][H] bf16
    bf16_t*       __restrict__ hall, long hall_stride)    // optional [B][T*H]
{
    __shared__ bf16_t Alds[2][64 * APAD];
    __shared__ bf16_t Blds[2][4][64 * APAD];

    const int tid   = threadIdx.x;
    const int lane  = tid & 31;
    const int wave  = tid >> 5;
    const int wgM   = blockIdx.x * 64;
    const int wgJ   = blockIdx.y * 64;
    const int waveM = (wave & 1) * 32;
    const int waveJ = (wave >> 1) * 32;

    v8f acc[4][2][2] = {};

    const int frow = tid >> 1;          // 0..63
    const int fcs  = (tid & 1) * 16;    // 0 or 16
    const int nk   = (KA + H_) >> 5;

    // Stage one 64x32 A tile + 4 gate 64x32 B tiles into LDS buffer `buf`.
    auto stage = [&](int buf, int kt) {
        {
            int k = kt + fcs;
            const bf16_t* src; long stride; int kk;
            if (k < KA) { src = xp;    stride = xstride; kk = k; }
            else        { src = hprev; stride = H_;      kk = k - KA; }
            const bf16_t* p = src + (long)(wgM + frow) * stride + kk;
            async_copy32(lds_addr_of(&Alds[buf][frow * APAD + fcs]), p);
        }
        for (int g = 0; g < 4; ++g) {
            int k = kt + fcs;
            const bf16_t* src; int stride; int kk;
            if (k < KA) { src = Wih; stride = KA; kk = k; }
            else        { src = Whh; stride = H_; kk = k - KA; }
            const bf16_t* p = src + (long)(g * H_ + wgJ + frow) * stride + kk;
            async_copy32(lds_addr_of(&Blds[buf][g][frow * APAD + fcs]), p);
        }
    };

    stage(0, 0);

    for (int ks = 0; ks < nk; ++ks) {
        const int cur = ks & 1;
        // Own async loads done -> barrier: all waves' tile[cur] writes visible,
        // and all waves are past reading tile[cur^1] from the prior step.
        wait_async0();
        __syncthreads();
        if (ks + 1 < nk) stage(cur ^ 1, (ks + 1) * 32);

        // ---- A fragments (16x32 bf16 layout, ISA 7.12.2) ----
        union { v16bf v; v8bf h[2]; } afr[2];
        {
            int am = lane & 15;
            int kh = (lane >> 4) * 8;   // lanes 0-15: K 0..7/16..23; 16-31: 8..15/24..31
            for (int mi = 0; mi < 2; ++mi) {
                const bf16_t* r = &Alds[cur][(waveM + mi * 16 + am) * APAD];
                afr[mi].h[0] = *(const v8bf*)(r + kh);
                afr[mi].h[1] = *(const v8bf*)(r + 16 + kh);
            }
        }
        // ---- per gate: B fragments (32x16) + 4 WMMAs ----
        {
            int bn  = lane & 15;
            int bkh = (lane >> 4) * 8;  // lanes 0-15: K 0..7/16..23; 16-31: 8..15/24..31
            for (int g = 0; g < 4; ++g) {
                union { v16bf v; v8bf h[2]; } bfr[2];
                for (int nj = 0; nj < 2; ++nj) {
                    const bf16_t* r = &Blds[cur][g][(waveJ + nj * 16 + bn) * APAD + bkh];
                    bfr[nj].h[0] = *(const v8bf*)r;
                    bfr[nj].h[1] = *(const v8bf*)(r + 16);
                }
                for (int mi = 0; mi < 2; ++mi)
                    for (int nj = 0; nj < 2; ++nj)
                        acc[g][mi][nj] = __builtin_amdgcn_wmma_f32_16x16x32_bf16(
                            false, afr[mi].v, false, bfr[nj].v,
                            (short)0, acc[g][mi][nj], false, false);
            }
        }
    }

    // ---- epilogue: biases, activations, cell update, h stores ----
    for (int mi = 0; mi < 2; ++mi) {
        for (int nj = 0; nj < 2; ++nj) {
            int j = wgJ + waveJ + nj * 16 + (lane & 15);
            float bi = bih[0 * H_ + j] + bhh[0 * H_ + j];
            float bf = bih[1 * H_ + j] + bhh[1 * H_ + j];
            float bg = bih[2 * H_ + j] + bhh[2 * H_ + j];
            float bo = bih[3 * H_ + j] + bhh[3 * H_ + j];
            int mbase = wgM + waveM + mi * 16 + (lane >> 4) * 8;
            for (int r = 0; r < 8; ++r) {
                int m = mbase + r;
                float gi = fast_sigmoid(acc[0][mi][nj][r] + bi);
                float gf = fast_sigmoid(acc[1][mi][nj][r] + bf);
                float gg = fast_tanh   (acc[2][mi][nj][r] + bg);
                float go = fast_sigmoid(acc[3][mi][nj][r] + bo);
                long ci = (long)m * H_ + j;
                float c = gf * cst[ci] + gi * gg;
                cst[ci] = c;
                float h = go * fast_tanh(c);
                hout[ci] = (bf16_t)h;
                if (hall) hall[(long)m * hall_stride + j] = (bf16_t)h;
            }
        }
    }
}

// out[m, n] = sum_k A[m,k]*W[n,k] + bias[n]   (M=256, N=128, K=512)
__global__ __launch_bounds__(128) void fc_kernel(
    const bf16_t* __restrict__ A,     // [256][512] bf16
    const bf16_t* __restrict__ W,     // [128][512] bf16
    const float*  __restrict__ bias,  // [128]
    float*        __restrict__ out)   // [256][128]
{
    __shared__ bf16_t Alds[2][64 * APAD];
    __shared__ bf16_t Blds[2][64 * APAD];

    const int tid   = threadIdx.x;
    const int lane  = tid & 31;
    const int wave  = tid >> 5;
    const int wgM   = blockIdx.x * 64;
    const int wgN   = blockIdx.y * 64;
    const int waveM = (wave & 1) * 32;
    const int waveN = (wave >> 1) * 32;

    v8f acc[2][2] = {};
    const int frow = tid >> 1;
    const int fcs  = (tid & 1) * 16;
    const int nk   = H_ >> 5;

    auto stage = [&](int buf, int kt) {
        const bf16_t* pa = A + (long)(wgM + frow) * H_ + kt + fcs;
        async_copy32(lds_addr_of(&Alds[buf][frow * APAD + fcs]), pa);
        const bf16_t* pb = W + (long)(wgN + frow) * H_ + kt + fcs;
        async_copy32(lds_addr_of(&Blds[buf][frow * APAD + fcs]), pb);
    };

    stage(0, 0);

    for (int ks = 0; ks < nk; ++ks) {
        const int cur = ks & 1;
        wait_async0();
        __syncthreads();
        if (ks + 1 < nk) stage(cur ^ 1, (ks + 1) * 32);

        union { v16bf v; v8bf h[2]; } afr[2], bfr[2];
        int am = lane & 15, kh = (lane >> 4) * 8;
        for (int mi = 0; mi < 2; ++mi) {
            const bf16_t* r = &Alds[cur][(waveM + mi * 16 + am) * APAD];
            afr[mi].h[0] = *(const v8bf*)(r + kh);
            afr[mi].h[1] = *(const v8bf*)(r + 16 + kh);
        }
        int bn = lane & 15, bkh = (lane >> 4) * 8;
        for (int nj = 0; nj < 2; ++nj) {
            const bf16_t* r = &Blds[cur][(waveN + nj * 16 + bn) * APAD + bkh];
            bfr[nj].h[0] = *(const v8bf*)r;
            bfr[nj].h[1] = *(const v8bf*)(r + 16);
        }
        for (int mi = 0; mi < 2; ++mi)
            for (int nj = 0; nj < 2; ++nj)
                acc[mi][nj] = __builtin_amdgcn_wmma_f32_16x16x32_bf16(
                    false, afr[mi].v, false, bfr[nj].v,
                    (short)0, acc[mi][nj], false, false);
    }

    for (int mi = 0; mi < 2; ++mi) {
        for (int nj = 0; nj < 2; ++nj) {
            int n = wgN + waveN + nj * 16 + (lane & 15);
            float b = bias[n];
            int mbase = wgM + waveM + mi * 16 + (lane >> 4) * 8;
            for (int r = 0; r < 8; ++r)
                out[(long)(mbase + r) * IN_ + n] = acc[mi][nj][r] + b;
        }
    }
}

extern "C" void kernel_launch(void* const* d_in, const int* in_sizes, int n_in,
                              void* d_out, int out_size, void* d_ws, size_t ws_size,
                              hipStream_t stream) {
    const float* x     = (const float*)d_in[0];
    const float* Wih0  = (const float*)d_in[1];
    const float* Whh0  = (const float*)d_in[2];
    const float* bih0  = (const float*)d_in[3];
    const float* bhh0  = (const float*)d_in[4];
    const float* Wih1  = (const float*)d_in[5];
    const float* Whh1  = (const float*)d_in[6];
    const float* bih1  = (const float*)d_in[7];
    const float* bhh1  = (const float*)d_in[8];
    const float* fc1w  = (const float*)d_in[9];
    const float* fc1b  = (const float*)d_in[10];
    float* out = (float*)d_out;

    // ---- workspace carve-out (256B aligned) ----
    char* base = (char*)d_ws;
    size_t off = 0;
    auto carve = [&](size_t bytes) -> char* {
        char* p = base + off;
        off += (bytes + 255) & ~(size_t)255;
        return p;
    };
    bf16_t* x_bf   = (bf16_t*)carve((size_t)B_ * T_ * IN_ * 2);
    bf16_t* wih0b  = (bf16_t*)carve((size_t)4 * H_ * IN_ * 2);
    bf16_t* whh0b  = (bf16_t*)carve((size_t)4 * H_ * H_ * 2);
    bf16_t* wih1b  = (bf16_t*)carve((size_t)4 * H_ * H_ * 2);
    bf16_t* whh1b  = (bf16_t*)carve((size_t)4 * H_ * H_ * 2);
    bf16_t* fcwb   = (bf16_t*)carve((size_t)IN_ * H_ * 2);
    bf16_t* hall0  = (bf16_t*)carve((size_t)B_ * T_ * H_ * 2);
    bf16_t* h0buf0 = (bf16_t*)carve((size_t)B_ * H_ * 2);
    bf16_t* h0buf1 = (bf16_t*)carve((size_t)B_ * H_ * 2);
    bf16_t* h1buf0 = (bf16_t*)carve((size_t)B_ * H_ * 2);
    bf16_t* h1buf1 = (bf16_t*)carve((size_t)B_ * H_ * 2);
    float*  c0     = (float*) carve((size_t)B_ * H_ * 4);
    float*  c1     = (float*) carve((size_t)B_ * H_ * 4);
    bf16_t* h0buf[2] = { h0buf0, h0buf1 };
    bf16_t* h1buf[2] = { h1buf0, h1buf1 };

    // ---- one-time bf16 conversions ----
    cvt_f32_to_bf16<<<2048, 256, 0, stream>>>(x,    x_bf,  B_ * T_ * IN_);
    cvt_f32_to_bf16<<<512,  256, 0, stream>>>(Wih0, wih0b, 4 * H_ * IN_);
    cvt_f32_to_bf16<<<1024, 256, 0, stream>>>(Whh0, whh0b, 4 * H_ * H_);
    cvt_f32_to_bf16<<<1024, 256, 0, stream>>>(Wih1, wih1b, 4 * H_ * H_);
    cvt_f32_to_bf16<<<1024, 256, 0, stream>>>(Whh1, whh1b, 4 * H_ * H_);
    cvt_f32_to_bf16<<<256,  256, 0, stream>>>(fc1w, fcwb,  IN_ * H_);

    // ---- layer 0 ----
    zero_u32<<<128, 256, 0, stream>>>((uint32_t*)h0buf0, B_ * H_ / 2);
    zero_u32<<<256, 256, 0, stream>>>((uint32_t*)c0,     B_ * H_);
    dim3 stepGrid(B_ / 64, H_ / 64);
    for (int t = 0; t < T_; ++t) {
        lstm_step_kernel<<<stepGrid, 128, 0, stream>>>(
            x_bf + (long)t * IN_, (long)T_ * IN_, IN_,
            h0buf[t & 1], wih0b, whh0b, bih0, bhh0,
            c0, h0buf[(t + 1) & 1],
            hall0 + (long)t * H_, (long)T_ * H_);
    }

    // ---- layer 1 ----
    zero_u32<<<128, 256, 0, stream>>>((uint32_t*)h1buf0, B_ * H_ / 2);
    zero_u32<<<256, 256, 0, stream>>>((uint32_t*)c1,     B_ * H_);
    for (int t = 0; t < T_; ++t) {
        lstm_step_kernel<<<stepGrid, 128, 0, stream>>>(
            hall0 + (long)t * H_, (long)T_ * H_, H_,
            h1buf[t & 1], wih1b, whh1b, bih1, bhh1,
            c1, h1buf[(t + 1) & 1],
            (bf16_t*)nullptr, 0);
    }

    // ---- final FC on last hidden state (T even -> lands in buffer 0) ----
    fc_kernel<<<dim3(B_ / 64, IN_ / 64), 128, 0, stream>>>(
        h1buf[T_ & 1], fcwb, fc1b, out);
}
